// BFeatContrastiveAuxGAT_43215960932951
// MI455X (gfx1250) — matrix operand order repaired and checked by reading, not compiled, not run on hardware
//
#include <hip/hip_runtime.h>

// ---------------- problem constants (fixed by the reference) ----------------
#define NN 1024           // nodes
#define EE 2048           // edges
#define DIM 512           // feature dim
#define HEADS 8
#define DKH 64            // per-head dim
#define NDEPTH 2
#define BTP 40            // LDS pitch (bf16 elems) of N-major B tile; 80B = 16B multiple

// ---------------- WMMA types ----------------
typedef __attribute__((ext_vector_type(16))) __bf16 v16bf;
typedef __attribute__((ext_vector_type(8)))  float  v8f;

__device__ __forceinline__ unsigned short f2bf(float f) {
  unsigned int u = __float_as_uint(f);
  u += 0x7fffu + ((u >> 16) & 1u);          // round-to-nearest-even
  return (unsigned short)(u >> 16);
}

__device__ __forceinline__ void atomicMaxF(float* addr, float val) {
  // IEEE ordering trick: int-max for >=0, uint-min for <0 (init is -inf)
  if (val >= 0.0f) atomicMax((int*)addr, __float_as_int(val));
  else             atomicMin((unsigned int*)addr, __float_as_uint(val));
}

// LDS-relative byte address of a __shared__ object (generic -> addrspace(3) cast,
// then truncate: async-to-LDS instructions take wave-LDS-relative addresses).
__device__ __forceinline__ unsigned lds_addr_of(const void* p) {
  return (unsigned)(unsigned long long)
         (__attribute__((address_space(3))) const char*)p;
}

// ======================================================================
// GEMM: A fp32 row-major MxK; B given *bf16 transposed* (N-major: Bt[N][K]).
// C = act(A*B + bias) fp32 (+ optional bf16 N-major companion Cbf).
// Block tile 128x64, 8 waves; wave tile 32x32 = 2 A-frags x 2 B-frags ->
// 4 WMMAs per K-step per wave (2.0 ds_load_b128 per WMMA).
// A tile staged fp32->bf16; B tile DMA'd with the CDNA5 async engine
// (one global_load_async_to_lds_b128 per thread; s_wait_asynccnt).
// Requires M%128==0, N%64==0, K%32==0.
// ======================================================================
__global__ __launch_bounds__(256) void k_gemm_nn(
    const float* __restrict__ A, int lda,
    const unsigned short* __restrict__ Bt, int ldbT,
    const float* __restrict__ bias,
    float* __restrict__ C, int ldc,
    unsigned short* __restrict__ Cbf, int ldcbf,
    int M, int K, int N, int relu)
{
  __shared__ unsigned short As[128][33];
  __shared__ alignas(16) unsigned short BsT[64][BTP];   // [n][k], n-major
  const int t = threadIdx.x;
  const int rowBase = blockIdx.y * 128;
  const int colBase = blockIdx.x * 64;
  const int lane = t & 31, wave = t >> 5;
  const int wm = wave & 3, wn = wave >> 2;       // 4x2 wave grid; 32x32 per wave
  const int lh = lane >> 4, ln = lane & 15;

  // per-thread async packet: 64 N-rows x 4 x 16B segs = 256 packets
  const int br = t >> 2, bseg = t & 3;
  const unsigned ldsB = lds_addr_of(&BsT[0][0]) + (unsigned)(br * (BTP * 2) + bseg * 16);

  v8f acc00 = {0,0,0,0,0,0,0,0}, acc01 = {0,0,0,0,0,0,0,0};
  v8f acc10 = {0,0,0,0,0,0,0,0}, acc11 = {0,0,0,0,0,0,0,0};

  for (int kt = 0; kt < K; kt += 32) {
    // kick off async DMA of the B tile into LDS (tracked by ASYNCcnt)
    {
      const unsigned short* gsrc = Bt + (size_t)(colBase + br) * ldbT + kt + bseg * 8;
      asm volatile("global_load_async_to_lds_b128 %0, %1, off"
                   :: "v"(ldsB), "v"((unsigned long long)gsrc) : "memory");
    }
    if (kt + 32 < K)   // gfx1250 prefetch of next A tile
      __builtin_prefetch(&A[(size_t)(rowBase + (t & 127)) * lda + kt + 32], 0, 3);
#pragma unroll
    for (int i = 0; i < 16; ++i) {               // stage A 128x32 (fp32 -> bf16)
      int e = t * 16 + i, r = e >> 5, c = e & 31;
      As[r][c] = f2bf(A[(size_t)(rowBase + r) * lda + kt + c]);
    }
    asm volatile("s_wait_asynccnt 0x0" ::: "memory");
    __syncthreads();
    union { v16bf v; unsigned short u[16]; } a0, a1, b0, b1;
#pragma unroll
    for (int j = 0; j < 16; ++j) {
      // A fragment per ISA 16-bit A 16x32 table (lane holds k = 8*lh+0..7 and 16+8*lh+0..7)
      int ka = ((j >> 3) << 4) + (lh << 3) + (((j >> 1) & 3) << 1) + (j & 1);
      a0.u[j] = As[wm * 32 + ln][ka];
      a1.u[j] = As[wm * 32 + 16 + ln][ka];
      // B fragment: lane column fixed; K in same table order as A
      int kb = ((j >> 3) << 4) + (lh << 3) + (j & 7);
      b0.u[j] = BsT[wn * 32 + ln][kb];
      b1.u[j] = BsT[wn * 32 + 16 + ln][kb];
    }
    acc00 = __builtin_amdgcn_wmma_f32_16x16x32_bf16(false, a0.v, false, b0.v,
                                                    (short)0, acc00, false, false);
    acc01 = __builtin_amdgcn_wmma_f32_16x16x32_bf16(false, a0.v, false, b1.v,
                                                    (short)0, acc01, false, false);
    acc10 = __builtin_amdgcn_wmma_f32_16x16x32_bf16(false, a1.v, false, b0.v,
                                                    (short)0, acc10, false, false);
    acc11 = __builtin_amdgcn_wmma_f32_16x16x32_bf16(false, a1.v, false, b1.v,
                                                    (short)0, acc11, false, false);
    __syncthreads();
  }
#pragma unroll
  for (int mi = 0; mi < 2; ++mi) {
    v8f va = mi ? acc10 : acc00;
    v8f vb = mi ? acc11 : acc01;
#pragma unroll
    for (int r = 0; r < 8; ++r) {                // C layout: m = r + 8*lh, n = ln
      int row = rowBase + wm * 32 + mi * 16 + (lh << 3) + r;
      int c0 = colBase + wn * 32 + ln, c1 = c0 + 16;
      float v0 = va[r] + (bias ? bias[c0] : 0.0f);
      float v1 = vb[r] + (bias ? bias[c1] : 0.0f);
      if (relu) { v0 = fmaxf(v0, 0.0f); v1 = fmaxf(v1, 0.0f); }
      C[(size_t)row * ldc + c0] = v0;
      C[(size_t)row * ldc + c1] = v1;
      if (Cbf) {                                 // bf16 companion, N-major
        Cbf[(size_t)c0 * ldcbf + row] = f2bf(v0);
        Cbf[(size_t)c1 * ldcbf + row] = f2bf(v1);
      }
    }
  }
}

// ======================================================================
// Scores GEMM: C = scale*(A * B^T) (+ dw bias) (masked). A,B fp32 activations,
// B is NxK row-major; key rows are K-contiguous so the N-major LDS tile is
// staged with straight row copies. Same 128x64 / 32x32-wave tiling.
// ======================================================================
__global__ __launch_bounds__(256) void k_gemm_nt_scores(
    const float* __restrict__ A, int lda,
    const float* __restrict__ B, int ldb,
    float* __restrict__ C, int ldc,
    int M, int K, int N, float scale,
    const float* __restrict__ dwb, const int* __restrict__ bids)
{
  __shared__ unsigned short As[128][33];
  __shared__ alignas(16) unsigned short BsT[64][BTP];
  const int t = threadIdx.x;
  const int rowBase = blockIdx.y * 128;
  const int colBase = blockIdx.x * 64;
  const int lane = t & 31, wave = t >> 5;
  const int wm = wave & 3, wn = wave >> 2;
  const int lh = lane >> 4, ln = lane & 15;

  v8f acc00 = {0,0,0,0,0,0,0,0}, acc01 = {0,0,0,0,0,0,0,0};
  v8f acc10 = {0,0,0,0,0,0,0,0}, acc11 = {0,0,0,0,0,0,0,0};

  for (int kt = 0; kt < K; kt += 32) {
#pragma unroll
    for (int i = 0; i < 16; ++i) {               // stage A 128x32
      int e = t * 16 + i, r = e >> 5, c = e & 31;
      As[r][c] = f2bf(A[(size_t)(rowBase + r) * lda + kt + c]);
    }
#pragma unroll
    for (int i = 0; i < 8; ++i) {                // key rows copied K-contiguous
      int e = t * 8 + i, n = e >> 5, c = e & 31;
      BsT[n][c] = f2bf(B[(size_t)(colBase + n) * ldb + kt + c]);
    }
    __syncthreads();
    union { v16bf v; unsigned short u[16]; } a0, a1, b0, b1;
#pragma unroll
    for (int j = 0; j < 16; ++j) {
      int ka = ((j >> 3) << 4) + (lh << 3) + (((j >> 1) & 3) << 1) + (j & 1);
      a0.u[j] = As[wm * 32 + ln][ka];
      a1.u[j] = As[wm * 32 + 16 + ln][ka];
      int kb = ((j >> 3) << 4) + (lh << 3) + (j & 7);
      b0.u[j] = BsT[wn * 32 + ln][kb];
      b1.u[j] = BsT[wn * 32 + 16 + ln][kb];
    }
    acc00 = __builtin_amdgcn_wmma_f32_16x16x32_bf16(false, a0.v, false, b0.v,
                                                    (short)0, acc00, false, false);
    acc01 = __builtin_amdgcn_wmma_f32_16x16x32_bf16(false, a0.v, false, b1.v,
                                                    (short)0, acc01, false, false);
    acc10 = __builtin_amdgcn_wmma_f32_16x16x32_bf16(false, a1.v, false, b0.v,
                                                    (short)0, acc10, false, false);
    acc11 = __builtin_amdgcn_wmma_f32_16x16x32_bf16(false, a1.v, false, b1.v,
                                                    (short)0, acc11, false, false);
    __syncthreads();
  }
#pragma unroll
  for (int mi = 0; mi < 2; ++mi) {
    v8f va = mi ? acc10 : acc00;
    v8f vb = mi ? acc11 : acc01;
#pragma unroll
    for (int r = 0; r < 8; ++r) {
      int row = rowBase + wm * 32 + mi * 16 + (lh << 3) + r;
      int c0 = colBase + wn * 32 + ln, c1 = c0 + 16;
      float v0 = va[r] * scale, v1 = vb[r] * scale;
      if (dwb) { v0 += dwb[(size_t)row * N + c0]; v1 += dwb[(size_t)row * N + c1]; }
      if (bids) {
        int bi = bids[row];
        if (bi != bids[c0]) v0 = -1e9f;
        if (bi != bids[c1]) v1 = -1e9f;
      }
      C[(size_t)row * ldc + c0] = v0;
      C[(size_t)row * ldc + c1] = v1;
    }
  }
}

// ---------------- row softmax (in place), one block per row ----------------
__global__ __launch_bounds__(256) void k_softmax_rows(float* __restrict__ S, int cols)
{
  __shared__ float red[256];
  float* row = S + (size_t)blockIdx.x * cols;
  int t = threadIdx.x;
  float mx = -3.0e38f;
  for (int c = t; c < cols; c += 256) mx = fmaxf(mx, row[c]);
  red[t] = mx; __syncthreads();
  for (int s = 128; s > 0; s >>= 1) { if (t < s) red[t] = fmaxf(red[t], red[t + s]); __syncthreads(); }
  mx = red[0]; __syncthreads();
  float sum = 0.0f;
  for (int c = t; c < cols; c += 256) { float e = __expf(row[c] - mx); row[c] = e; sum += e; }
  red[t] = sum; __syncthreads();
  for (int s = 128; s > 0; s >>= 1) { if (t < s) red[t] += red[t + s]; __syncthreads(); }
  float inv = 1.0f / red[0];
  for (int c = t; c < cols; c += 256) row[c] *= inv;
}

// ---------------- layernorm over D=512 with optional residual ----------------
__global__ __launch_bounds__(256) void k_layernorm512(
    const float* __restrict__ X, const float* __restrict__ R,
    const float* __restrict__ g, const float* __restrict__ b,
    float* __restrict__ Y)
{
  __shared__ float red[256];
  int t = threadIdx.x;
  size_t base = (size_t)blockIdx.x * 512;
  float v0 = X[base + t], v1 = X[base + t + 256];
  if (R) { v0 += R[base + t]; v1 += R[base + t + 256]; }
  red[t] = v0 + v1; __syncthreads();
  for (int s = 128; s > 0; s >>= 1) { if (t < s) red[t] += red[t + s]; __syncthreads(); }
  float mean = red[0] * (1.0f / 512.0f); __syncthreads();
  float d0 = v0 - mean, d1 = v1 - mean;
  red[t] = d0 * d0 + d1 * d1; __syncthreads();
  for (int s = 128; s > 0; s >>= 1) { if (t < s) red[t] += red[t + s]; __syncthreads(); }
  float inv = rsqrtf(red[0] * (1.0f / 512.0f) + 1e-5f);
  Y[base + t]       = d0 * inv * g[t]       + b[t];
  Y[base + t + 256] = d1 * inv * g[t + 256] + b[t + 256];
}

// ---------------- distance-bias MLP: one thread per (i,j) pair ----------------
__global__ __launch_bounds__(256) void k_distbias(
    const float* __restrict__ center, const int* __restrict__ bids,
    const float* __restrict__ w0, const float* __restrict__ b0,
    const float* __restrict__ w1, const float* __restrict__ b1,
    const float* __restrict__ w2, const float* __restrict__ b2,
    const float* __restrict__ g0, const float* __restrict__ be0,
    const float* __restrict__ g1, const float* __restrict__ be1,
    float* __restrict__ dw)
{
  int p = blockIdx.x * blockDim.x + threadIdx.x;
  int i = p >> 10, j = p & 1023;
  float dx = center[j * 3 + 0] - center[i * 3 + 0];
  float dy = center[j * 3 + 1] - center[i * 3 + 1];
  float dz = center[j * 3 + 2] - center[i * 3 + 2];
  float in4[4] = { dx, dy, dz, sqrtf(dx * dx + dy * dy + dz * dz) };
  float h[32]; float mean = 0.0f;
  for (int o = 0; o < 32; ++o) {
    float s = b0[o];
    for (int k = 0; k < 4; ++k) s += in4[k] * w0[k * 32 + o];
    s = fmaxf(s, 0.0f); h[o] = s; mean += s;
  }
  mean *= (1.0f / 32.0f);
  float var = 0.0f;
  for (int o = 0; o < 32; ++o) { float d = h[o] - mean; var += d * d; }
  float inv = rsqrtf(var * (1.0f / 32.0f) + 1e-5f);
  for (int o = 0; o < 32; ++o) h[o] = (h[o] - mean) * inv * g0[o] + be0[o];
  float h2[32]; mean = 0.0f;
  for (int o = 0; o < 32; ++o) {
    float s = b1[o];
    for (int k = 0; k < 32; ++k) s += h[k] * w1[k * 32 + o];
    s = fmaxf(s, 0.0f); h2[o] = s; mean += s;
  }
  mean *= (1.0f / 32.0f); var = 0.0f;
  for (int o = 0; o < 32; ++o) { float d = h2[o] - mean; var += d * d; }
  inv = rsqrtf(var * (1.0f / 32.0f) + 1e-5f);
  for (int o = 0; o < 32; ++o) h2[o] = (h2[o] - mean) * inv * g1[o] + be1[o];
  int same = (bids[i] == bids[j]);
  for (int hd = 0; hd < 8; ++hd) {
    float s = b2[hd];
    for (int k = 0; k < 32; ++k) s += h2[k] * w2[k * 8 + hd];
    dw[((size_t)hd << 20) + ((size_t)i << 10) + j] = same ? s : 0.0f;
  }
}

// ---------------- GAT helper kernels ----------------
__global__ void k_gather_cat3(const float* __restrict__ x, const float* __restrict__ ef,
                              const int* __restrict__ src, const int* __restrict__ tgt,
                              float* __restrict__ o)
{
  int idx = blockIdx.x * blockDim.x + threadIdx.x;   // EE*1536
  int e = idx / 1536, d = idx - e * 1536;
  float v;
  if (d < 512)        v = x[(size_t)src[e] * 512 + d];
  else if (d < 1024)  v = ef[(size_t)e * 512 + (d - 512)];
  else                v = x[(size_t)tgt[e] * 512 + (d - 1024)];
  o[idx] = v;
}

__global__ void k_build_f(const float* __restrict__ pqx, const float* __restrict__ pee,
                          const int* __restrict__ src, float* __restrict__ f)
{
  int idx = blockIdx.x * blockDim.x + threadIdx.x;   // EE*HEADS*128
  int c = idx & 127, eh = idx >> 7, e = eh >> 3, h = eh & 7;
  float v;
  if (c < 64) v = pqx[(size_t)src[e] * DIM + c * HEADS + h];
  else        v = pee[(size_t)e * DIM + (c - 64) * HEADS + h];
  f[idx] = v;
}

__global__ void k_deg(const int* __restrict__ src, int* __restrict__ deg)
{
  int e = blockIdx.x * blockDim.x + threadIdx.x;
  if (e < EE) atomicAdd(&deg[src[e]], 1);
}

__global__ __launch_bounds__(256) void k_gate_scatter(
    const float* __restrict__ prob, const float* __restrict__ pvx,
    const int* __restrict__ src, const int* __restrict__ tgt,
    float* __restrict__ agg)
{
  int eh = blockIdx.x * blockDim.x + threadIdx.x;    // EE*HEADS
  int e = eh >> 3, h = eh & 7;
  const float* pr = prob + (size_t)eh * 64;
  float mx = -3.0e38f;
  for (int o = 0; o < 64; ++o) mx = fmaxf(mx, pr[o]);
  float ex[64]; float s = 0.0f;
  for (int o = 0; o < 64; ++o) { ex[o] = __expf(pr[o] - mx); s += ex[o]; }
  float inv = 1.0f / s;
  int tj = tgt[e], si = src[e];
  for (int o = 0; o < 64; ++o) {
    float m = ex[o] * inv * pvx[(size_t)tj * DIM + o * HEADS + h];
    atomicMaxF(&agg[(size_t)si * DIM + o * HEADS + h], m);
  }
}

__global__ void k_fix_agg(float* __restrict__ agg, const int* __restrict__ deg)
{
  int i = blockIdx.x * blockDim.x + threadIdx.x;     // NN*DIM
  if (deg[i >> 9] == 0) agg[i] = 0.0f;
}

__global__ void k_cat2(const float* __restrict__ x, const float* __restrict__ agg,
                       float* __restrict__ o)
{
  int idx = blockIdx.x * blockDim.x + threadIdx.x;   // NN*1024
  int n = idx >> 10, d = idx & 1023;
  o[idx] = (d < 512) ? x[(size_t)n * 512 + d] : agg[(size_t)n * 512 + (d - 512)];
}

// ---------------- tiny elementwise kernels ----------------
__global__ void k_copy(const float* __restrict__ a, float* __restrict__ o, int n)
{ int i = blockIdx.x * blockDim.x + threadIdx.x; if (i < n) o[i] = a[i]; }
__global__ void k_relu_ip(float* __restrict__ a, int n)
{ int i = blockIdx.x * blockDim.x + threadIdx.x; if (i < n) a[i] = fmaxf(a[i], 0.0f); }
__global__ void k_fill(float* __restrict__ a, float v, int n)
{ int i = blockIdx.x * blockDim.x + threadIdx.x; if (i < n) a[i] = v; }
__global__ void k_izero(int* __restrict__ a, int n)
{ int i = blockIdx.x * blockDim.x + threadIdx.x; if (i < n) a[i] = 0; }
// convert fp32 KxN weight to bf16 NxK (transposed, for N-major async B tiles)
__global__ void k_f2bfT(const float* __restrict__ w, unsigned short* __restrict__ o,
                        int K, int N)
{
  int idx = blockIdx.x * blockDim.x + threadIdx.x;   // K*N
  if (idx >= K * N) return;
  int n = idx / K, k = idx - n * K;
  o[idx] = f2bf(w[(size_t)k * N + n]);
}

// ======================================================================
// Host orchestration
// ======================================================================
extern "C" void kernel_launch(void* const* d_in, const int* in_sizes, int n_in,
                              void* d_out, int out_size, void* d_ws, size_t ws_size,
                              hipStream_t stream)
{
  (void)in_sizes; (void)n_in; (void)out_size; (void)ws_size;

  const float* in_xs  = (const float*)d_in[0];
  const float* in_xc  = (const float*)d_in[1];
  const float* in_es  = (const float*)d_in[2];
  const float* in_ec  = (const float*)d_in[3];
  const int*   eidx   = (const int*)d_in[4];
  const int*   bids   = (const int*)d_in[5];
  const float* center = (const float*)d_in[6];
  const int* src = eidx;
  const int* tgt = eidx + EE;

  // ---- decode params (jax pytree leaf order: dict keys sorted, lists in order)
  struct LinP { const float* w; const float* b; const unsigned short* wbf; };
  struct MhaP { LinP q, k, v, o; const float* g; const float* beta; };
  struct GatP { LinP ne0, ne1, nn0, nn1, pe, pq, pr0, pr1, pv; };
  int pi = 7;
  auto nf    = [&]() { return (const float*)d_in[pi++]; };
  auto rdLin = [&](LinP& l) { l.b = nf(); l.w = nf(); l.wbf = nullptr; };  // keys: b, w
  auto rdMha = [&](MhaP& m) {                                  // keys: b,g,k,o,q,v
    m.beta = nf(); m.g = nf();
    rdLin(m.k); rdLin(m.o); rdLin(m.q); rdLin(m.v);
  };
  auto rdGat = [&](GatP& gg) {                                 // keys sorted
    rdLin(gg.ne0); rdLin(gg.ne1); rdLin(gg.nn0); rdLin(gg.nn1);
    rdLin(gg.pe);  rdLin(gg.pq);  rdLin(gg.pr0); rdLin(gg.pr1); rdLin(gg.pv);
  };
  // top-level sorted: cross_attn, cross_attn_rel, fc0, fc1, fc2,
  //                   gcn_con, gcn_sgg, ln0, ln1, self_attn
  MhaP crossA[NDEPTH], crossR[NDEPTH], selfA[NDEPTH];
  GatP gcnC[NDEPTH], gcnS[NDEPTH];
  LinP fc0, fc1, fc2;
  rdMha(crossA[0]); rdMha(crossA[1]);
  rdMha(crossR[0]); rdMha(crossR[1]);
  rdLin(fc0); rdLin(fc1); rdLin(fc2);
  rdGat(gcnC[0]); rdGat(gcnC[1]);
  rdGat(gcnS[0]); rdGat(gcnS[1]);
  const float* ln0b = nf(); const float* ln0g = nf();
  const float* ln1b = nf(); const float* ln1g = nf();
  rdMha(selfA[0]); rdMha(selfA[1]);

  // ---- workspace partition (floats)
  float* W = (float*)d_ws;
  size_t off = 0;
  auto alloc = [&](size_t n) { float* p = W + off; off += n; return p; };
  const size_t U = (size_t)EE * DIM;                   // 1M floats
  float* dwb = alloc((size_t)HEADS * NN * NN);         // (H,N,N) bias, 8M
  float* S   = alloc((size_t)EE * EE);                 // per-head scores, 4M
  float* P[6]; for (int i = 0; i < 6; ++i) P[i] = alloc(U);
  float* xsA = alloc((size_t)NN * DIM); float* xsB = alloc((size_t)NN * DIM);
  float* xcA = alloc((size_t)NN * DIM); float* xcB = alloc((size_t)NN * DIM);
  float* esA = alloc(U);                float* esB = alloc(U);
  float* ecA = alloc(U);                float* ecB = alloc(U);
  int* deg = (int*)alloc(NN);
  unsigned short* vbfT = (unsigned short*)alloc(U / 2 + 64);  // bf16 V, N-major DIMxRk

  // ---- pre-convert all GEMM weights to bf16, transposed to NxK (N-major)
  auto cvt = [&](LinP& l, int K, int N) {
    size_t ne = (size_t)K * N;
    unsigned short* o = (unsigned short*)alloc(ne / 2 + 64);
    k_f2bfT<<<(int)((ne + 255) / 256), 256, 0, stream>>>(l.w, o, K, N);
    l.wbf = o;
  };
  for (int i = 0; i < NDEPTH; ++i) {
    MhaP* ms[3] = { &selfA[i], &crossA[i], &crossR[i] };
    for (int m = 0; m < 3; ++m) {
      cvt(ms[m]->q, DIM, DIM); cvt(ms[m]->k, DIM, DIM);
      cvt(ms[m]->v, DIM, DIM); cvt(ms[m]->o, DIM, DIM);
    }
    GatP* gs[2] = { &gcnS[i], &gcnC[i] };
    for (int g = 0; g < 2; ++g) {
      cvt(gs[g]->ne0, 3 * DIM, 2 * DIM);
      cvt(gs[g]->ne1, 2 * DIM, DIM);
      cvt(gs[g]->nn0, 128, 128);
      cvt(gs[g]->nn1, 128, 64);
      cvt(gs[g]->pe,  DIM, DIM);
      cvt(gs[g]->pq,  DIM, DIM);
      cvt(gs[g]->pv,  DIM, DIM);
      cvt(gs[g]->pr0, 2 * DIM, 2 * DIM);
      cvt(gs[g]->pr1, 2 * DIM, DIM);
    }
  }

  // gemm: Bt is N-major bf16 with row length ldbT (== K of the weight)
  auto gemm = [&](const float* A, int lda, const unsigned short* Bt, int ldbT,
                  const float* bias, float* C, int ldc,
                  unsigned short* Cbf, int ldcbf,
                  int M, int K, int Ncol, int relu) {
    dim3 g(Ncol / 64, M / 128);
    k_gemm_nn<<<g, 256, 0, stream>>>(A, lda, Bt, ldbT, bias, C, ldc,
                                     Cbf, ldcbf, M, K, Ncol, relu);
  };
  auto swp = [](float*& a, float*& b) { float* t = a; a = b; b = t; };

  auto run_mha = [&](const MhaP& p, const float* qin, const float* kin,
                     int Rq, int Rk, const float* dwbias, const int* mask,
                     float* dst) {
    float *qb = P[0], *kb = P[1], *vb = P[2], *ao = P[3], *pj = P[4];
    gemm(qin, DIM, p.q.wbf, DIM, p.q.b, qb, DIM, nullptr, 0, Rq, DIM, DIM, 0);
    gemm(kin, DIM, p.k.wbf, DIM, p.k.b, kb, DIM, nullptr, 0, Rk, DIM, DIM, 0);
    gemm(kin, DIM, p.v.wbf, DIM, p.v.b, vb, DIM, vbfT, Rk, Rk, DIM, DIM, 0);
    for (int h = 0; h < HEADS; ++h) {
      dim3 g(Rk / 64, Rq / 128);
      k_gemm_nt_scores<<<g, 256, 0, stream>>>(
          qb + h * DKH, DIM, kb + h * DKH, DIM, S, Rk, Rq, DKH, Rk, 0.125f,
          dwbias ? dwbias + (size_t)h * NN * NN : nullptr, mask);
      k_softmax_rows<<<Rq, 256, 0, stream>>>(S, Rk);
      // PV: B operand = rows h*64..h*64+63 of the N-major V companion
      gemm(S, Rk, vbfT + (size_t)(h * DKH) * Rk, Rk, nullptr,
           ao + h * DKH, DIM, nullptr, 0, Rq, Rk, DKH, 0);
    }
    gemm(ao, DIM, p.o.wbf, DIM, p.o.b, pj, DIM, nullptr, 0, Rq, DIM, DIM, 0);
    k_layernorm512<<<Rq, 256, 0, stream>>>(pj, qin, p.g, p.beta, dst);
  };

  auto run_gat = [&](const GatP& p, const float* x, const float* e,
                     float* xout, float* eout) {
    float* cat3 = P[0];                                 // 3 units
    float* hid1 = P[3];                                 // 2 units
    k_gather_cat3<<<(EE * 1536) / 256, 256, 0, stream>>>(x, e, src, tgt, cat3);
    gemm(cat3, 3 * DIM, p.ne0.wbf, 3 * DIM, p.ne0.b, hid1, 2 * DIM, nullptr, 0,
         EE, 3 * DIM, 2 * DIM, 1);
    gemm(hid1, 2 * DIM, p.ne1.wbf, 2 * DIM, p.ne1.b, eout, DIM, nullptr, 0,
         EE, 2 * DIM, DIM, 0);
    float* pqx = P[0]; float* pee = P[2]; float* pvx = P[5];
    gemm(x, DIM, p.pq.wbf, DIM, p.pq.b, pqx, DIM, nullptr, 0, NN, DIM, DIM, 0);
    gemm(x, DIM, p.pv.wbf, DIM, p.pv.b, pvx, DIM, nullptr, 0, NN, DIM, DIM, 0);
    gemm(e, DIM, p.pe.wbf, DIM, p.pe.b, pee, DIM, nullptr, 0, EE, DIM, DIM, 0);
    float* fbuf = P[3];                                 // 2 units
    k_build_f<<<(EE * HEADS * 128) / 256, 256, 0, stream>>>(pqx, pee, src, fbuf);
    float* hid2 = P[0];                                 // 2 units
    gemm(fbuf, 128, p.nn0.wbf, 128, p.nn0.b, hid2, 128, nullptr, 0,
         EE * HEADS, 128, 128, 1);
    float* prob = P[2];
    gemm(hid2, 128, p.nn1.wbf, 128, p.nn1.b, prob, 64, nullptr, 0,
         EE * HEADS, 128, 64, 0);
    float* agg = P[3];
    k_fill<<<(NN * DIM) / 256, 256, 0, stream>>>(agg, -3.0e38f, NN * DIM);
    k_izero<<<4, 256, 0, stream>>>(deg, NN);
    k_deg<<<EE / 256, 256, 0, stream>>>(src, deg);
    k_gate_scatter<<<(EE * HEADS) / 256, 256, 0, stream>>>(prob, pvx, src, tgt, agg);
    k_fix_agg<<<(NN * DIM) / 256, 256, 0, stream>>>(agg, deg);
    float* cat2 = P[4];
    k_cat2<<<(NN * 1024) / 256, 256, 0, stream>>>(x, agg, cat2);
    float* h3 = P[0];
    gemm(cat2, 2 * DIM, p.pr0.wbf, 2 * DIM, p.pr0.b, h3, 2 * DIM, nullptr, 0,
         NN, 2 * DIM, 2 * DIM, 1);
    gemm(h3, 2 * DIM, p.pr1.wbf, 2 * DIM, p.pr1.b, xout, DIM, nullptr, 0,
         NN, 2 * DIM, DIM, 0);
  };

  // ---- init persistent activations
  k_copy<<<(NN * DIM) / 256, 256, 0, stream>>>(in_xs, xsA, NN * DIM);
  k_copy<<<(NN * DIM) / 256, 256, 0, stream>>>(in_xc, xcA, NN * DIM);
  k_copy<<<(int)U / 256, 256, 0, stream>>>(in_es, esA, (int)U);
  k_copy<<<(int)U / 256, 256, 0, stream>>>(in_ec, ecA, (int)U);
  float *xs = xsA, *xsSp = xsB, *xc = xcA, *xcSp = xcB;
  float *es = esA, *esSp = esB, *ec = ecA, *ecSp = ecB;

  // ---- distance bias (H,N,N), computed once
  k_distbias<<<(NN * NN) / 256, 256, 0, stream>>>(
      center, bids, fc0.w, fc0.b, fc1.w, fc1.b, fc2.w, fc2.b,
      ln0g, ln0b, ln1g, ln1b, dwb);

  // ---- GNN depth loop
  for (int i = 0; i < NDEPTH; ++i) {
    run_mha(selfA[i], xs, xs, NN, NN, dwb, bids, xsSp);            // xs_b
    run_mha(crossA[i], xc, xsSp, NN, NN, dwb, bids, xcSp);         // xc <- cross(xc, xs_b)
    swp(xc, xcSp);
    swp(xs, xsSp);                                                  // xs = xs_b
    run_gat(gcnS[i], xs, es, xsSp, esSp); swp(xs, xsSp); swp(es, esSp);
    run_gat(gcnC[i], xc, ec, xcSp, ecSp); swp(xc, xcSp); swp(ec, ecSp);
    run_mha(crossR[i], ec, es, EE, EE, nullptr, nullptr, ecSp);    // ec <- rel(ec, es)
    swp(ec, ecSp);
    if (i < NDEPTH - 1) {
      k_relu_ip<<<(NN * DIM) / 256, 256, 0, stream>>>(xs, NN * DIM);
      k_relu_ip<<<(NN * DIM) / 256, 256, 0, stream>>>(xc, NN * DIM);
      k_relu_ip<<<(int)U / 256, 256, 0, stream>>>(es, (int)U);
      k_relu_ip<<<(int)U / 256, 256, 0, stream>>>(ec, (int)U);
    }
  }

  // ---- pack outputs: [xs, xc, es, ec]
  float* out = (float*)d_out;
  k_copy<<<(NN * DIM) / 256, 256, 0, stream>>>(xs, out, NN * DIM);
  k_copy<<<(NN * DIM) / 256, 256, 0, stream>>>(xc, out + (size_t)NN * DIM, NN * DIM);
  k_copy<<<(int)U / 256, 256, 0, stream>>>(es, out + 2 * (size_t)NN * DIM, (int)U);
  k_copy<<<(int)U / 256, 256, 0, stream>>>(ec, out + 2 * (size_t)NN * DIM + U, (int)U);
}
